// VectorQuantizerEMA_30872224923701
// MI455X (gfx1250) — matrix operand order repaired and checked by reading, not compiled, not run on hardware
//
#include <hip/hip_runtime.h>
#include <math.h>

#define EMB_DIM 256
#define NUM_EMB 8192
#define NTOK    32768
#define NQ      (NTOK * EMB_DIM)   // 8388608 floats of quantized output
#define BETA    0.25f
#define M_BLK   64                 // tokens per block (4 wave-tiles x 16)
#define KT      32                 // codes staged per iteration
#define BSTR    264                // 256 + 8 bf16 pad; rows stay 16B aligned

typedef __attribute__((ext_vector_type(16))) __bf16 v16bf;
typedef __attribute__((ext_vector_type(4)))  __bf16 v4bf;
typedef __attribute__((ext_vector_type(8)))  float  v8f;

// ---------- zero counts + scalar output slots ----------
__global__ void vq_init(float* counts, float* scal) {
  int gid = blockIdx.x * blockDim.x + threadIdx.x;
  if (gid < NUM_EMB) counts[gid] = 0.0f;
  if (gid < 3) scal[gid] = 0.0f;
}

// ---------- ||e_k||^2 per code ----------
__global__ void vq_norms(const float* __restrict__ emb, float* __restrict__ nrm) {
  int k = blockIdx.x * blockDim.x + threadIdx.x;   // coalesced across k
  float s = 0.f;
#pragma unroll 8
  for (int d = 0; d < EMB_DIM; ++d) {
    float e = emb[(size_t)d * NUM_EMB + k];
    s += e * e;
  }
  nrm[k] = s;
}

// ---------- main: WMMA distances + argmin + quantize + loss ----------
__global__ __launch_bounds__(256) void vq_main(
    const float* __restrict__ z, const float* __restrict__ emb,
    const float* __restrict__ nrm, float* __restrict__ counts,
    float* __restrict__ out) {
  __shared__ __align__(16) __bf16 sB[2][KT * BSTR];  // staged codebook tiles [code][d]
  __shared__ float sBV[8][16];
  __shared__ int   sBI[8][16];
  __shared__ int   sBest[M_BLK];
  __shared__ float sRed[8];

  const int tid  = threadIdx.x;
  const int lane = tid & 31;
  const int wave = tid >> 5;
  const int half = lane >> 4;
  const int l15  = lane & 15;
  const int mt   = wave & 3;       // this wave's 16-token tile
  const int ks   = wave >> 2;      // this wave's 16-code half of the staged tile
  const int tb   = blockIdx.x * M_BLK;

  // ---- A fragments (this wave's 16 tokens, all of D) held in registers: 64 VGPRs
  v16bf a[8];
  {
    const float* zrow = z + (size_t)(tb + mt * 16 + l15) * EMB_DIM;
#pragma unroll
    for (int ds = 0; ds < 8; ++ds) {
      v16bf t;
#pragma unroll
      for (int e = 0; e < 16; ++e) {
        int d = ds * 32 + (e & 7) + 8 * half + 16 * (e >> 3);  // 16-bit A layout
        t[e] = (__bf16)zrow[d];
      }
      a[ds] = t;
    }
  }

  // ---- cooperative stage of one 32-code tile into sB[buf] as bf16 [code][d]
  auto stage = [&](int kb2, int buf) {
    const int j   = tid & 31;                 // code within tile (coalesced dim)
    const int db0 = tid >> 5;                 // d-block (4 dims) base
    const float* col = emb + kb2 + j;
#pragma unroll
    for (int i = 0; i < 8; ++i) {
      int d = (db0 + 8 * i) * 4;              // 0..252
      float x0 = col[(size_t)(d + 0) * NUM_EMB];
      float x1 = col[(size_t)(d + 1) * NUM_EMB];
      float x2 = col[(size_t)(d + 2) * NUM_EMB];
      float x3 = col[(size_t)(d + 3) * NUM_EMB];
      v4bf h = { (__bf16)x0, (__bf16)x1, (__bf16)x2, (__bf16)x3 };
      *(v4bf*)&sB[buf][j * BSTR + d] = h;     // 8B store
    }
  };

  float rbV[8];
  int   rbI[8];
#pragma unroll
  for (int r = 0; r < 8; ++r) { rbV[r] = -3.402823466e38f; rbI[r] = 0; }

  stage(0, 0);
  __syncthreads();

  union BF { uint4 u[2]; v16bf v; };

  for (int kb = 0; kb < NUM_EMB; kb += KT) {
    const int p = (kb >> 5) & 1;
    if (kb + KT < NUM_EMB) stage(kb + KT, p ^ 1);   // pipeline next tile

    const int   kcol = kb + ks * 16 + l15;          // this lane's code (B "n")
    const float nv   = nrm[kcol];

    const __bf16* bbase = &sB[p][(ks * 16 + l15) * BSTR];

    // B-fragment register double-buffer: load frag ds+1 before WMMA on frag ds,
    // so the compiler can wait on dscnt<=2 and overlap LDS latency with WMMA.
    BF f0, f1;
    f0.u[0] = *(const uint4*)(bbase + 8 * half);
    f0.u[1] = *(const uint4*)(bbase + 8 * half + 16);

    v8f c = {};
#pragma unroll
    for (int ds = 0; ds < 8; ++ds) {
      BF& cur = (ds & 1) ? f1 : f0;
      BF& nxt = (ds & 1) ? f0 : f1;
      if (ds < 7) {
        const __bf16* bp = bbase + (ds + 1) * 32 + 8 * half;
        nxt.u[0] = *(const uint4*)(bp);      // elements 0..7
        nxt.u[1] = *(const uint4*)(bp + 16); // elements 8..15
      }
      c = __builtin_amdgcn_wmma_f32_16x16x32_bf16(
              false, a[ds], false, cur.v, (short)0, c, false, false);
    }
    // score = 2*s - ||e||^2 ; k increases monotonically -> '>' keeps first argmin
#pragma unroll
    for (int r = 0; r < 8; ++r) {
      float sc = __builtin_fmaf(2.0f, c[r], -nv);
      if (sc > rbV[r]) { rbV[r] = sc; rbI[r] = kcol; }
    }
    __syncthreads();
  }

  // ---- one cross-lane butterfly per r, then per-wave LDS write
#pragma unroll
  for (int r = 0; r < 8; ++r) {
    float sc = rbV[r];
    int   ki = rbI[r];
#pragma unroll
    for (int mask = 1; mask <= 8; mask <<= 1) {
      float osc = __shfl_xor(sc, mask, 32);
      int   oki = __shfl_xor(ki, mask, 32);
      if (osc > sc || (osc == sc && oki < ki)) { sc = osc; ki = oki; }
    }
    if (l15 == 0) {                       // lanes 0 & 16 -> rows r and r+8
      sBV[wave][r + 8 * half] = sc;
      sBI[wave][r + 8 * half] = ki;
    }
  }
  __syncthreads();

  // ---- merge the two k-halves (waves mt and mt+4), emit indices + counts
  if (tid < M_BLK) {
    int mtt = tid >> 4, r16 = tid & 15;
    float v0 = sBV[mtt][r16];     int i0 = sBI[mtt][r16];
    float v1 = sBV[mtt + 4][r16]; int i1 = sBI[mtt + 4][r16];
    int bi = (v1 > v0 || (v1 == v0 && i1 < i0)) ? i1 : i0;
    sBest[tid] = bi;
    out[NQ + tb + tid] = (float)bi;
    atomicAdd(&counts[bi], 1.0f);
  }
  __syncthreads();

  // ---- quantized output (forward value of straight-through) + commitment loss
  float lsum = 0.f;
  {
    const float4* zb = (const float4*)(z + (size_t)tb * EMB_DIM);
    float4*       ob = (float4*)(out + (size_t)tb * EMB_DIM);
#pragma unroll
    for (int j = 0; j < 16; ++j) {
      int chunk = tid + 256 * j;
      int m  = chunk >> 6;
      int d  = (chunk & 63) * 4;
      int kq = sBest[m];
      const float* ecol = emb + kq;
      float4 q;
      q.x = ecol[(size_t)(d + 0) * NUM_EMB];
      q.y = ecol[(size_t)(d + 1) * NUM_EMB];
      q.z = ecol[(size_t)(d + 2) * NUM_EMB];
      q.w = ecol[(size_t)(d + 3) * NUM_EMB];
      float4 zv = zb[chunk];
      float dx = q.x - zv.x, dy = q.y - zv.y, dz = q.z - zv.z, dw = q.w - zv.w;
      lsum += dx * dx + dy * dy + dz * dz + dw * dw;
      ob[chunk] = q;
    }
  }
#pragma unroll
  for (int mask = 1; mask < 32; mask <<= 1) lsum += __shfl_xor(lsum, mask, 32);
  if (lane == 0) sRed[wave] = lsum;
  __syncthreads();
  if (tid == 0) {
    float t = 0.f;
    for (int w = 0; w < 8; ++w) t += sRed[w];
    atomicAdd(&out[NQ + NTOK + 1], t * (BETA / (float)NQ));
  }
}

// ---------- perplexity ----------
__global__ void vq_finalize(const float* __restrict__ counts, float* __restrict__ out) {
  __shared__ float sRed[8];
  int tid = threadIdx.x;
  float s = 0.f;
  for (int k = tid; k < NUM_EMB; k += 256) {
    float p = counts[k] * (1.0f / (float)NTOK);
    s -= p * logf(p + 1e-10f);
  }
#pragma unroll
  for (int mask = 1; mask < 32; mask <<= 1) s += __shfl_xor(s, mask, 32);
  if ((tid & 31) == 0) sRed[tid >> 5] = s;
  __syncthreads();
  if (tid == 0) {
    float t = 0.f;
    for (int w = 0; w < 8; ++w) t += sRed[w];
    out[NQ + NTOK + 0] = 0.0f;      // literal 0 output
    out[NQ + NTOK + 2] = expf(t);   // perplexity
  }
}

extern "C" void kernel_launch(void* const* d_in, const int* in_sizes, int n_in,
                              void* d_out, int out_size, void* d_ws, size_t ws_size,
                              hipStream_t stream) {
  const float* z   = (const float*)d_in[0];   // [32*32*32, 256] f32
  const float* emb = (const float*)d_in[1];   // [256, 8192] f32
  float* out    = (float*)d_out;              // quantized | indices | 0 | loss | perplexity
  float* nrm    = (float*)d_ws;               // 8192 f32
  float* counts = nrm + NUM_EMB;              // 8192 f32

  vq_init    <<<NUM_EMB / 256, 256, 0, stream>>>(counts, out + NQ + NTOK);
  vq_norms   <<<NUM_EMB / 256, 256, 0, stream>>>(emb, nrm);
  vq_main    <<<NTOK / M_BLK, 256, 0, stream>>>(z, emb, nrm, counts, out);
  vq_finalize<<<1, 256, 0, stream>>>(counts, out);
}
